// SparseBlock_35673998361274
// MI455X (gfx1250) — matrix operand order repaired and checked by reading, not compiled, not run on hardware
//
#include <hip/hip_runtime.h>

// ---------------------------------------------------------------------------
// Block-sparse 3x3 conv for MI455X (gfx1250), bf16 WMMA path.
// Geometry fixed by the reference:
#define HW     1024
#define CIN    64
#define COUT   64
#define BCD    32          // blocks per spatial dim
#define BS     34          // gathered block size (input tile)
#define BOUT   32          // output block size
#define NTAP   9           // 3x3 taps
#define XPIX   (BS * BS)   // 1156 pixels per input tile
#define XSTR   72          // padded bf16 stride per pixel in LDS (144B, 16B-mult)
#define WSTR   72          // padded bf16 stride per weight row in LDS

typedef __attribute__((ext_vector_type(16))) __bf16 v16bf;
typedef __attribute__((ext_vector_type(8)))  __bf16 v8bf;
typedef __attribute__((ext_vector_type(8)))  float  v8f;

// Hard fence: no instruction may be scheduled across (bounds the live-fragment
// window per 3x3 tap => VGPR allocation stays <= 256), plus an IR-level memory
// clobber so LICM can't hoist the loop-invariant B fragments either.
__device__ __forceinline__ void sched_fence() {
    asm volatile("" ::: "memory");
#if defined(__has_builtin)
#if __has_builtin(__builtin_amdgcn_sched_barrier)
    __builtin_amdgcn_sched_barrier(0);
#endif
#endif
}

// Within one tap window: pin the pipeline shape "24 DS-reads, then 16 WMMAs"
// so every WMMA has maximal ds_load lookahead instead of load->wait(0)->wmma.
__device__ __forceinline__ void sched_tap_groups() {
#if defined(__has_builtin)
#if __has_builtin(__builtin_amdgcn_sched_group_barrier)
    __builtin_amdgcn_sched_group_barrier(0x100, 24, 0); // DS read group
    __builtin_amdgcn_sched_group_barrier(0x008, 16, 0); // MFMA/WMMA group
#endif
#endif
}

// A-fragment: per ISA 16-bit A layout, lane needs two 16B chunks 32B apart.
__device__ __forceinline__ v16bf ld_pair32(const __bf16* p) {
    union { v16bf v; v8bf h[2]; } u;
    u.h[0] = *reinterpret_cast<const v8bf*>(p);
    u.h[1] = *reinterpret_cast<const v8bf*>(p + 16);
    return u.v;
}
// B-fragment: per ISA 16-bit B layout, lane needs two 16B chunks 32B apart.
__device__ __forceinline__ v16bf ld_cont(const __bf16* p) {
    union { v16bf v; v8bf h[2]; } u;
    u.h[0] = *reinterpret_cast<const v8bf*>(p);
    u.h[1] = *reinterpret_cast<const v8bf*>(p + 16);
    return u.v;
}

// --- prep: zero active-block flags; convert+transpose weights to bf16 -------
// Wk is [ky][kx][cin][cout] fp32 -> wbf is [tap][cout][cin] bf16.
__global__ void prep_kernel(const float* __restrict__ Wk,
                            int* __restrict__ flags,
                            unsigned short* __restrict__ wbf_u) {
    __bf16* wbf = reinterpret_cast<__bf16*>(wbf_u);
    const int tid = blockIdx.x * blockDim.x + threadIdx.x;
    if (tid < BCD * BCD) flags[tid] = 0;
    if (tid < NTAP * CIN * COUT) {
        const int tap = tid / (CIN * COUT);
        const int rem = tid % (CIN * COUT);
        const int co  = rem / CIN;
        const int ci  = rem % CIN;
        wbf[(tap * COUT + co) * CIN + ci] = (__bf16)Wk[(tap * CIN + ci) * COUT + co];
    }
}

// --- mark active blocks -----------------------------------------------------
__global__ void mark_kernel(const int* __restrict__ ids, int n,
                            int* __restrict__ flags) {
    const int tid = blockIdx.x * blockDim.x + threadIdx.x;
    if (tid < n) flags[ids[tid]] = 1;
}

// --- zero only inactive output blocks (halves zero-fill traffic) ------------
__global__ void zero_inactive_kernel(const int* __restrict__ flags,
                                     float* __restrict__ out) {
    const int b = blockIdx.x;
    if (flags[b]) return;
    const int ii = b >> 5, jj = b & 31;
    const float4 z = make_float4(0.f, 0.f, 0.f, 0.f);
    for (int p = threadIdx.x; p < BOUT * BOUT; p += blockDim.x) {
        const int oy = p >> 5, ox = p & 31;
        float4* dst = reinterpret_cast<float4*>(
            out + ((size_t)(ii * BOUT + oy) * HW + (jj * BOUT + ox)) * COUT);
        #pragma unroll
        for (int j = 0; j < 16; ++j) dst[j] = z;
    }
}

// --- main conv: one workgroup (8 wave32) per active block -------------------
__global__ void __launch_bounds__(256)
conv_kernel(const float* __restrict__ x,
            const int* __restrict__ ids,
            const unsigned short* __restrict__ wbf_u,
            const float* __restrict__ bias,
            float* __restrict__ out) {
    __shared__ __align__(16) __bf16 xs[XPIX * XSTR];         // 166,464 B
    __shared__ __align__(16) __bf16 wsm[NTAP * COUT * WSTR]; //  82,944 B
    __shared__ float biass[COUT];                            //     256 B

    const __bf16* wbf = reinterpret_cast<const __bf16*>(wbf_u);
    const int tid = threadIdx.x;
    const int bid = ids[blockIdx.x];
    const int ii = (bid >> 5) & 31;
    const int jj = bid & 31;

    // Stage bf16 weights into LDS (padded rows).
    for (int row = tid; row < NTAP * COUT; row += 256) {
        const v8bf* s = reinterpret_cast<const v8bf*>(wbf + row * CIN);
        v8bf* d = reinterpret_cast<v8bf*>(wsm + row * WSTR);
        #pragma unroll
        for (int j = 0; j < 8; ++j) d[j] = s[j];
    }
    if (tid < COUT) biass[tid] = bias[tid];

    // Stage 34x34x64 input tile (with implicit pad=1) fp32 -> bf16 into LDS.
    const int r0  = ii * BOUT - 1;   // top-left of tile in unpadded coords
    const int cg0 = jj * BOUT - 1;
    for (int p = tid; p < XPIX; p += 256) {
        const int pr = p / BS, pc = p - pr * BS;
        const int gr = r0 + pr, gc = cg0 + pc;
        v8bf* dst = reinterpret_cast<v8bf*>(xs + p * XSTR);
        if ((unsigned)gr < HW && (unsigned)gc < HW) {
            const float4* src = reinterpret_cast<const float4*>(
                x + ((size_t)gr * HW + gc) * CIN);
            #pragma unroll
            for (int j = 0; j < 8; ++j) {
                const float4 f0 = src[2 * j], f1 = src[2 * j + 1];
                v8bf v;
                v[0] = (__bf16)f0.x; v[1] = (__bf16)f0.y;
                v[2] = (__bf16)f0.z; v[3] = (__bf16)f0.w;
                v[4] = (__bf16)f1.x; v[5] = (__bf16)f1.y;
                v[6] = (__bf16)f1.z; v[7] = (__bf16)f1.w;
                dst[j] = v;
            }
        } else {
            v8bf z;
            #pragma unroll
            for (int i = 0; i < 8; ++i) z[i] = (__bf16)0.f;
            #pragma unroll
            for (int j = 0; j < 8; ++j) dst[j] = z;
        }
    }
    __syncthreads();

    const int wv = tid >> 5;       // wave id 0..7
    const int lane = tid & 31;
    const int lm = lane & 15;      // A: row M / B,D: col N
    const int lh = lane >> 4;      // lane half selects K sub-range

    float bn[4];
    #pragma unroll
    for (int t = 0; t < 4; ++t) bn[t] = biass[t * 16 + lm];

    // Each wave: 8 M-tiles (16 pixels each) as 4 pairs; acc block = 2x4 tiles.
    #pragma unroll 1
    for (int pr2 = 0; pr2 < 4; ++pr2) {
        const int mt0  = wv * 8 + pr2 * 2;
        const int pix0 = mt0 * 16 + lm;
        const int oy0 = pix0 >> 5, ox0 = pix0 & 31;
        const int pix1 = pix0 + 16;
        const int oy1 = pix1 >> 5, ox1 = pix1 & 31;
        const int ab0 = (oy0 * BS + ox0) * XSTR + 8 * lh;
        const int ab1 = (oy1 * BS + ox1) * XSTR + 8 * lh;
        const int wb  = lm * WSTR + 8 * lh;

        v8f acc[2][4];
        #pragma unroll
        for (int m = 0; m < 2; ++m) {
            #pragma unroll
            for (int n = 0; n < 4; ++n) {
                #pragma unroll
                for (int i = 0; i < 8; ++i) acc[m][n][i] = 0.f;
            }
        }

        #pragma unroll
        for (int ky = 0; ky < 3; ++ky) {
            #pragma unroll
            for (int kx = 0; kx < 3; ++kx) {
                sched_fence();   // tap-boundary pressure cap
                const int tap = ky * 3 + kx;
                const int sh  = (ky * BS + kx) * XSTR;

                // All 24 DS reads of this tap first...
                v16bf A0[2], A1[2], Bf[2][4];
                #pragma unroll
                for (int hf = 0; hf < 2; ++hf) {
                    const int cb = hf * 32;
                    A0[hf] = ld_pair32(xs + ab0 + sh + cb);
                    A1[hf] = ld_pair32(xs + ab1 + sh + cb);
                    const __bf16* wp = wsm + tap * COUT * WSTR + wb + cb;
                    #pragma unroll
                    for (int nt = 0; nt < 4; ++nt)
                        Bf[hf][nt] = ld_cont(wp + nt * 16 * WSTR);
                }

                sched_tap_groups();  // pin: 24 DS reads, then 16 WMMAs

                // ...then the 16 WMMAs draining them.
                #pragma unroll
                for (int hf = 0; hf < 2; ++hf) {
                    #pragma unroll
                    for (int nt = 0; nt < 4; ++nt) {
                        acc[0][nt] = __builtin_amdgcn_wmma_f32_16x16x32_bf16(
                            false, A0[hf], false, Bf[hf][nt], (short)0,
                            acc[0][nt], false, false);
                        acc[1][nt] = __builtin_amdgcn_wmma_f32_16x16x32_bf16(
                            false, A1[hf], false, Bf[hf][nt], (short)0,
                            acc[1][nt], false, false);
                    }
                }
            }
        }

        // Scatter: +bias, direct store into the block's output region.
        #pragma unroll
        for (int m = 0; m < 2; ++m) {
            #pragma unroll
            for (int r = 0; r < 8; ++r) {
                const int pix = (mt0 + m) * 16 + 8 * lh + r; // D: M = r + 8*lh
                const int oy = pix >> 5, ox = pix & 31;
                float* drow = out +
                    ((size_t)(ii * BOUT + oy) * HW + (jj * BOUT + ox)) * COUT + lm;
                #pragma unroll
                for (int nt = 0; nt < 4; ++nt)
                    drow[nt * 16] = acc[m][nt][r] + bn[nt];
            }
        }
    }
}

// ---------------------------------------------------------------------------
extern "C" void kernel_launch(void* const* d_in, const int* in_sizes, int n_in,
                              void* d_out, int out_size, void* d_ws, size_t ws_size,
                              hipStream_t stream) {
    const float* x    = (const float*)d_in[0];
    const int*   ids  = (const int*)d_in[1];
    const float* Wk   = (const float*)d_in[2];
    const float* bias = (const float*)d_in[3];
    float* out = (float*)d_out;
    const int nact = in_sizes[1];

    // Workspace layout: [0,4K) flags (1024 ints); [8K, 8K+72KB) bf16 weights.
    int* flags = (int*)d_ws;
    unsigned short* wbf = (unsigned short*)((char*)d_ws + 8192);

    prep_kernel<<<(NTAP * CIN * COUT + 255) / 256, 256, 0, stream>>>(Wk, flags, wbf);
    mark_kernel<<<(nact + 255) / 256, 256, 0, stream>>>(ids, nact, flags);
    zero_inactive_kernel<<<BCD * BCD, 256, 0, stream>>>(flags, out);
    conv_kernel<<<nact, 256, 0, stream>>>(x, ids, wbf, bias, out);
}
